// TransformerEncoder_40372692582539
// MI455X (gfx1250) — matrix operand order repaired and checked by reading, not compiled, not run on hardware
//
#include <hip/hip_runtime.h>
#include <hip/hip_bf16.h>

typedef __attribute__((ext_vector_type(16))) _Float16 v16h;
typedef __attribute__((ext_vector_type(8)))  float    v8f;

// ---------------- WMMA wrapper (guarded so host pass compiles) --------------
__device__ __forceinline__ v8f wmma_f32_f16(v16h a, v16h b, v8f c) {
#if defined(__gfx1250__)
  return __builtin_amdgcn_wmma_f32_16x16x32_f16(false, a, false, b, (short)0, c,
                                                false, false);
#else
  return c;
#endif
}

// -------- CDNA5 async global->LDS (ASYNCcnt-tracked) ------------------------
// LDS operand = low 32 bits of the generic pointer (flat aperture rule).
__device__ __forceinline__ void async_b128(void* lds, const void* g) {
#if defined(__gfx1250__)
  const unsigned loff = (unsigned)(size_t)lds;
  asm volatile("global_load_async_to_lds_b128 %0, %1, off"
               :: "v"(loff), "v"(g) : "memory");
#else
  *reinterpret_cast<uint4*>(lds) = *reinterpret_cast<const uint4*>(g);
#endif
}
__device__ __forceinline__ void async_wait0() {
#if defined(__gfx1250__)
  asm volatile("s_wait_asynccnt 0x0" ::: "memory");
#endif
}
// Async loads complete in order: <=3 outstanding -> the previous tile's 3
// per-thread loads are done, only the prefetched tile's 3 remain in flight.
__device__ __forceinline__ void async_wait_le3() {
#if defined(__gfx1250__)
  asm volatile("s_wait_asynccnt 0x3" ::: "memory");
#endif
}

// A-matrix fragment (16x32 f16): lane = M row (mod 16); lanes 0-15 carry
// K = {0..7,16..23}, lanes 16-31 carry K = {8..15,24..31} (VGPR j = K pair).
__device__ __forceinline__ v16h frag_a(const _Float16* base, int row0, int ld,
                                       int koff) {
  const int l = threadIdx.x & 31;
  const _Float16* p = base + (row0 + (l & 15)) * ld + koff + ((l & 16) ? 8 : 0);
  v16h f;
#pragma unroll
  for (int j = 0; j < 4; ++j) {
    f[2 * j]     = p[2 * j];
    f[2 * j + 1] = p[2 * j + 1];
    f[8 + 2 * j]     = p[16 + 2 * j];
    f[8 + 2 * j + 1] = p[16 + 2 * j + 1];
  }
  return f;
}

// B-matrix fragment (32x16 f16, row-major [k][n] in LDS): lane = N column (mod 16);
// lanes 0-15 carry K = {0..7,16..23}, lanes 16-31 carry K = {8..15,24..31}.
__device__ __forceinline__ v16h frag_b(const _Float16* base, int n0, int ld,
                                       int koff) {
  const int l = threadIdx.x & 31;
  const _Float16* p = base + (koff + ((l & 16) ? 8 : 0)) * ld + n0 + (l & 15);
  v16h f;
#pragma unroll
  for (int i = 0; i < 8; ++i) {
    f[i] = p[i * ld];
    f[8 + i] = p[(16 + i) * ld];
  }
  return f;
}

// ---------------- GEMM: C[M,N] = A[M,K] * B[K,N], tiled 128x64x32 -----------
// Double-buffered async-to-LDS pipeline: prefetch tile k+1 while WMMAs run on
// tile k.  MODE 0: f16+bias (QKV)  1: f16+bias+GELU (MLP up)
//          MODE 2: f32+bias+resid (O-proj, MLP down)
template <int MODE>
__global__ __launch_bounds__(256) void gemm_wmma(
    const _Float16* __restrict__ A, const _Float16* __restrict__ Bw,
    const float* __restrict__ bias, const float* __restrict__ resid,
    void* __restrict__ outp, int M, int N, int K) {
  __shared__ __align__(16) _Float16 sA[2][128 * 32];
  __shared__ __align__(16) _Float16 sB[2][32 * 64];
  const int tid = threadIdx.x;
  const int wave = tid >> 5;
  const int wm = wave & 3, wn = wave >> 2;
  const int rowBase = blockIdx.y * 128;
  const int colBase = blockIdx.x * 64;

  const int ar = tid >> 1, aseg = (tid & 1) * 16;   // A: 128 rows x 32 halves
  const int bkr = tid >> 3, bseg = (tid & 7) * 8;   // B: 32 rows x 64 halves

  auto stage = [&](int buf, int k0) {               // 3 async b128 per thread
    async_b128(&sA[buf][ar * 32 + aseg],
               A + (size_t)(rowBase + ar) * K + k0 + aseg);
    async_b128(&sA[buf][ar * 32 + aseg + 8],
               A + (size_t)(rowBase + ar) * K + k0 + aseg + 8);
    async_b128(&sB[buf][bkr * 64 + bseg],
               Bw + (size_t)(k0 + bkr) * N + colBase + bseg);
  };

  v8f acc[2][2] = {};
  stage(0, 0);
  for (int k0 = 0, it = 0; k0 < K; k0 += 32, ++it) {
    const int cur = it & 1;
    if (k0 + 32 < K) {
      stage(cur ^ 1, k0 + 32);  // prefetch next tile into the other buffer
      async_wait_le3();         // current tile's loads are complete
    } else {
      async_wait0();
    }
    __syncthreads();
#pragma unroll
    for (int mi = 0; mi < 2; ++mi) {
      v16h a = frag_a(sA[cur], wm * 32 + mi * 16, 32, 0);
#pragma unroll
      for (int ni = 0; ni < 2; ++ni) {
        v16h b = frag_b(sB[cur], wn * 32 + ni * 16, 64, 0);
        acc[mi][ni] = wmma_f32_f16(a, b, acc[mi][ni]);
      }
    }
    __syncthreads();
  }
  // Epilogue: C layout — VGPR r holds row r (+8 for upper half-wave), col = lane%16
  const int l = tid & 31;
  const int nloc = l & 15;
  const int rhalf = (l & 16) ? 8 : 0;
#pragma unroll
  for (int mi = 0; mi < 2; ++mi)
#pragma unroll
    for (int ni = 0; ni < 2; ++ni)
#pragma unroll
      for (int r = 0; r < 8; ++r) {
        int gr = rowBase + wm * 32 + mi * 16 + r + rhalf;
        int gc = colBase + wn * 32 + ni * 16 + nloc;
        float v = acc[mi][ni][r] + bias[gc];
        if (MODE == 1) v = 0.5f * v * (1.0f + erff(v * 0.70710678118654752f));
        size_t idx = (size_t)gr * N + gc;
        if (MODE == 2)
          ((float*)outp)[idx] = v + resid[idx];
        else
          ((_Float16*)outp)[idx] = (_Float16)v;
      }
}

// ---------------- Flash attention: block = (b, h, 128 q rows) ---------------
__global__ __launch_bounds__(256) void attn_wmma(
    const _Float16* __restrict__ Q, const _Float16* __restrict__ Kg,
    const _Float16* __restrict__ Vg, _Float16* __restrict__ ctx, int S, int D,
    int H) {
  const int h = blockIdx.y, b = blockIdx.z;
  const int qBase = blockIdx.x * 128;
  const int tid = threadIdx.x, wave = tid >> 5, l = tid & 31;
  __shared__ __align__(16) _Float16 sQ[128 * 64];    // [qrow][d]
  __shared__ __align__(16) _Float16 sKt[64 * 64];    // [d][key]  (B of QK^T)
  __shared__ __align__(16) _Float16 sV[64 * 64];     // [key][d]  (B of PV)
  __shared__ __align__(16) _Float16 sP[8 * 16 * 64]; // per-wave P reshuffle

  {  // stage Q tile 128x64 via async engine (4 x b128 per thread)
    int r = tid >> 1, seg = (tid & 1) * 32;
    const _Float16* g = Q + (size_t)(b * S + qBase + r) * D + h * 64 + seg;
    _Float16* s = sQ + r * 64 + seg;
#pragma unroll
    for (int i = 0; i < 4; ++i) async_b128(s + i * 8, g + i * 8);
  }

  float mrow[8], lrow[8];
  v8f oacc[4] = {};
#pragma unroll
  for (int r = 0; r < 8; ++r) { mrow[r] = -1e30f; lrow[r] = 0.f; }
  const float scale = 0.125f;  // 1/sqrt(64)
  const int rhalf = (l & 16) ? 8 : 0;
  const int pbase = wave * 1024;

  for (int kt = 0; kt < S / 64; ++kt) {
    __syncthreads();
    {  // stage K transposed (ds stores) + V direct (async b128)
      int kr = tid >> 2, dseg = (tid & 3) * 16;
      const _Float16* gk =
          Kg + (size_t)(b * S + kt * 64 + kr) * D + h * 64 + dseg;
#pragma unroll
      for (int i = 0; i < 16; ++i) sKt[(dseg + i) * 64 + kr] = gk[i];
      const _Float16* gv =
          Vg + (size_t)(b * S + kt * 64 + kr) * D + h * 64 + dseg;
      _Float16* sv = sV + kr * 64 + dseg;
      async_b128(sv, gv);
      async_b128(sv + 8, gv + 8);
    }
    async_wait0();  // covers V (and first-iteration Q) async loads
    __syncthreads();

    // S = Q(16x64) x K^T(64x64) for this wave's 16 rows
    v16h aq0 = frag_a(sQ, wave * 16, 64, 0);
    v16h aq1 = frag_a(sQ, wave * 16, 64, 32);
    v8f sc[4];
#pragma unroll
    for (int nt = 0; nt < 4; ++nt) {
      v8f s = {};
      s = wmma_f32_f16(aq0, frag_b(sKt, nt * 16, 64, 0), s);
      s = wmma_f32_f16(aq1, frag_b(sKt, nt * 16, 64, 32), s);
      sc[nt] = s;
    }

    // Online softmax: per-lane reg r holds row (r + rhalf); reduce over the
    // 4 column tiles then across the 16 lanes of the half-wave.
    float mnew[8], alpha[8];
#pragma unroll
    for (int r = 0; r < 8; ++r) {
      float mx = fmaxf(fmaxf(sc[0][r], sc[1][r]), fmaxf(sc[2][r], sc[3][r]));
      mx *= scale;
#pragma unroll
      for (int off = 1; off < 16; off <<= 1)
        mx = fmaxf(mx, __shfl_xor(mx, off, 16));
      mnew[r] = fmaxf(mrow[r], mx);
      alpha[r] = __expf(mrow[r] - mnew[r]);
    }
#pragma unroll
    for (int r = 0; r < 8; ++r) {
      float sum = 0.f;
#pragma unroll
      for (int nt = 0; nt < 4; ++nt) {
        float p = __expf(sc[nt][r] * scale - mnew[r]);
        sum += p;
        sP[pbase + (r + rhalf) * 64 + nt * 16 + (l & 15)] = (_Float16)p;
      }
#pragma unroll
      for (int off = 1; off < 16; off <<= 1) sum += __shfl_xor(sum, off, 16);
      lrow[r] = lrow[r] * alpha[r] + sum;
      mrow[r] = mnew[r];
#pragma unroll
      for (int dt = 0; dt < 4; ++dt) oacc[dt][r] *= alpha[r];
    }

    // O += P(16x64) x V(64x64) — P re-read from LDS in A-fragment layout
    v16h ap0 = frag_a(sP + pbase, 0, 64, 0);
    v16h ap1 = frag_a(sP + pbase, 0, 64, 32);
#pragma unroll
    for (int dt = 0; dt < 4; ++dt) {
      oacc[dt] = wmma_f32_f16(ap0, frag_b(sV, dt * 16, 64, 0), oacc[dt]);
      oacc[dt] = wmma_f32_f16(ap1, frag_b(sV, dt * 16, 64, 32), oacc[dt]);
    }
  }

#pragma unroll
  for (int dt = 0; dt < 4; ++dt)
#pragma unroll
    for (int r = 0; r < 8; ++r) {
      int row = qBase + wave * 16 + r + rhalf;
      float v = oacc[dt][r] / lrow[r];
      ctx[(size_t)(b * S + row) * D + h * 64 + dt * 16 + (l & 15)] =
          (_Float16)v;
    }
}

// ---------------- LayerNorm (f32 in, f16 out), one block per row ------------
__global__ __launch_bounds__(256) void ln_f16(const float* __restrict__ x,
                                              const float* __restrict__ g,
                                              const float* __restrict__ be,
                                              _Float16* __restrict__ out) {
  const int row = blockIdx.x, tid = threadIdx.x;
  __shared__ float r1[256], r2[256];
  float xv[3], s = 0.f, sq = 0.f;
#pragma unroll
  for (int i = 0; i < 3; ++i) {
    float v = x[(size_t)row * 768 + tid + i * 256];
    xv[i] = v; s += v; sq += v * v;
  }
  r1[tid] = s; r2[tid] = sq;
  __syncthreads();
  for (int off = 128; off > 0; off >>= 1) {
    if (tid < off) { r1[tid] += r1[tid + off]; r2[tid] += r2[tid + off]; }
    __syncthreads();
  }
  float mu = r1[0] * (1.0f / 768.0f);
  float var = r2[0] * (1.0f / 768.0f) - mu * mu;
  float rs = rsqrtf(var + 1e-6f);
#pragma unroll
  for (int i = 0; i < 3; ++i) {
    int c = tid + i * 256;
    out[(size_t)row * 768 + c] = (_Float16)((xv[i] - mu) * rs * g[c] + be[c]);
  }
}

__global__ void cast_f16(const float* __restrict__ in,
                         _Float16* __restrict__ out, int n) {
  int i = blockIdx.x * blockDim.x + threadIdx.x;
  if (i < n) out[i] = (_Float16)in[i];
}

// ---------------------------------------------------------------------------
extern "C" void kernel_launch(void* const* d_in, const int* in_sizes, int n_in,
                              void* d_out, int out_size, void* d_ws,
                              size_t ws_size, hipStream_t stream) {
  (void)in_sizes; (void)n_in; (void)out_size; (void)ws_size;
  const int B = 8, S = 1024, D = 768, H = 12, F = 3072;
  const int M = B * S;

  const float* x   = (const float*)d_in[0];
  const float* Wq  = (const float*)d_in[1];
  const float* bq  = (const float*)d_in[2];
  const float* Wk  = (const float*)d_in[3];
  const float* bk  = (const float*)d_in[4];
  const float* Wv  = (const float*)d_in[5];
  const float* bv  = (const float*)d_in[6];
  const float* Wo  = (const float*)d_in[7];
  const float* bo  = (const float*)d_in[8];
  const float* g1  = (const float*)d_in[9];
  const float* be1 = (const float*)d_in[10];
  const float* g2  = (const float*)d_in[11];
  const float* be2 = (const float*)d_in[12];
  const float* W1  = (const float*)d_in[13];
  const float* b1  = (const float*)d_in[14];
  const float* W2  = (const float*)d_in[15];
  const float* b2  = (const float*)d_in[16];

  char* w = (char*)d_ws;
  size_t off = 0;
  auto carve = [&](size_t bytes) {
    char* p = w + off;
    off += (bytes + 255) & ~(size_t)255;
    return p;
  };
  _Float16* hq  = (_Float16*)carve((size_t)M * D * 2);
  _Float16* WqH = (_Float16*)carve((size_t)D * D * 2);
  _Float16* WkH = (_Float16*)carve((size_t)D * D * 2);
  _Float16* WvH = (_Float16*)carve((size_t)D * D * 2);
  _Float16* WoH = (_Float16*)carve((size_t)D * D * 2);
  _Float16* W1H = (_Float16*)carve((size_t)D * F * 2);
  _Float16* W2H = (_Float16*)carve((size_t)F * D * 2);
  _Float16* Qh  = (_Float16*)carve((size_t)M * D * 2);
  _Float16* Kh  = (_Float16*)carve((size_t)M * D * 2);
  _Float16* Vh  = (_Float16*)carve((size_t)M * D * 2);
  _Float16* ctx = (_Float16*)carve((size_t)M * D * 2);
  float*    x2  = (float*)   carve((size_t)M * D * 4);
  _Float16* y1  = (_Float16*)carve((size_t)M * D * 2);
  _Float16* mid = (_Float16*)carve((size_t)M * F * 2);

  // Pre-LN + weight casts
  ln_f16<<<M, 256, 0, stream>>>(x, g1, be1, hq);
  cast_f16<<<(D * D + 255) / 256, 256, 0, stream>>>(Wq, WqH, D * D);
  cast_f16<<<(D * D + 255) / 256, 256, 0, stream>>>(Wk, WkH, D * D);
  cast_f16<<<(D * D + 255) / 256, 256, 0, stream>>>(Wv, WvH, D * D);
  cast_f16<<<(D * D + 255) / 256, 256, 0, stream>>>(Wo, WoH, D * D);
  cast_f16<<<(D * F + 255) / 256, 256, 0, stream>>>(W1, W1H, D * F);
  cast_f16<<<(F * D + 255) / 256, 256, 0, stream>>>(W2, W2H, F * D);

  // QKV projections
  dim3 gDD(D / 64, M / 128);
  gemm_wmma<0><<<gDD, 256, 0, stream>>>(hq, WqH, bq, nullptr, Qh, M, D, D);
  gemm_wmma<0><<<gDD, 256, 0, stream>>>(hq, WkH, bk, nullptr, Kh, M, D, D);
  gemm_wmma<0><<<gDD, 256, 0, stream>>>(hq, WvH, bv, nullptr, Vh, M, D, D);

  // Flash attention
  attn_wmma<<<dim3(S / 128, H, B), 256, 0, stream>>>(Qh, Kh, Vh, ctx, S, D, H);

  // O-projection + residual -> x2 (f32)
  gemm_wmma<2><<<gDD, 256, 0, stream>>>(ctx, WoH, bo, x, x2, M, D, D);

  // MLP
  ln_f16<<<M, 256, 0, stream>>>(x2, g2, be2, y1);
  gemm_wmma<1><<<dim3(F / 64, M / 128), 256, 0, stream>>>(y1, W1H, b1, nullptr,
                                                          mid, M, F, D);
  gemm_wmma<2><<<gDD, 256, 0, stream>>>(mid, W2H, b2, x2, (float*)d_out, M, D,
                                        F);
}
